// GQAAttention_25709674234134
// MI455X (gfx1250) — matrix operand order repaired and checked by reading, not compiled, not run on hardware
//
#include <hip/hip_runtime.h>

typedef unsigned short ushort_t;
typedef __bf16 v16bf __attribute__((ext_vector_type(16)));
typedef float  v8f   __attribute__((ext_vector_type(8)));

#define D_MODEL 2048
#define N_HEAD  16
#define N_KV    4
#define D_HEAD  128
#define BATCH   2
#define SEQ     2048
#define ROWS    (BATCH*SEQ)     /* 4096 */
#define KV_DIM  (N_KV*D_HEAD)   /* 512  */

// ---------------- helpers ----------------

__device__ __forceinline__ ushort_t f2bf(float f){
  union { float f; unsigned u; } x; x.f = f;
  unsigned r = x.u + 0x7fffu + ((x.u >> 16) & 1u);   // round-to-nearest-even
  return (ushort_t)(r >> 16);
}

__device__ __forceinline__ v8f wmma_bf16(v16bf a, v16bf b, v8f c){
  // D = A(16x32 bf16) * B(32x16 bf16) + C(16x16 f32)
  return __builtin_amdgcn_wmma_f32_16x16x32_bf16(false, a, false, b,
                                                 (short)0, c, false, false);
}

// A-operand fragment (16x32, MxK). rowptr = start of this lane's row (caller
// picks row = lane&15). Layout: lanes<16 carry K{0..15} of each
// 32-chunk, lanes>=16 carry K{16..31}.
__device__ __forceinline__ v16bf ldfragA(const ushort_t* rowptr, int kbase, int lane){
  union { v16bf v; unsigned u[8]; } f;
  const int hi = (lane & 16) ? 16 : 0;
  const unsigned* q0 = (const unsigned*)(rowptr + kbase + hi);
  const unsigned* q1 = (const unsigned*)(rowptr + kbase + 8 + hi);
  f.u[0]=q0[0]; f.u[1]=q0[1]; f.u[2]=q0[2]; f.u[3]=q0[3];
  f.u[4]=q1[0]; f.u[5]=q1[1]; f.u[6]=q1[2]; f.u[7]=q1[3];
  return f.v;
}

// B-operand fragment (32x16, KxN), column-contiguous storage. colptr = start
// of this lane's column (caller picks col = lane&15). Layout (as ldfragA): lanes 0-15
// hold K=0..15, lanes 16-31 hold K=16..31, consecutive in the 8 VGPRs.
__device__ __forceinline__ v16bf ldfragB(const ushort_t* colptr, int kbase, int lane){
  union { v16bf v; unsigned u[8]; } f;
  const int hi = (lane & 16) ? 16 : 0;
  const unsigned* q0 = (const unsigned*)(colptr + kbase + hi);
  #pragma unroll
  for (int i = 0; i < 8; ++i) f.u[i] = q0[i];
  return f.v;
}

// ---------------- conversion / packing kernels ----------------

__global__ void f32_to_bf16_k(const float* __restrict__ in, ushort_t* __restrict__ out, int n){
  int i = blockIdx.x*blockDim.x + threadIdx.x;
  if (i < n) out[i] = f2bf(in[i]);
}

// in: fp32 [K][N] row-major  ->  out: bf16 [N][K] (transposed)
__global__ void transpose_to_bf16(const float* __restrict__ in, ushort_t* __restrict__ out,
                                  int K, int N){
  int i = blockIdx.x*blockDim.x + threadIdx.x;
  if (i < K*N){
    int kk = i / N, nn = i - kk*N;
    out[(size_t)nn*K + kk] = f2bf(in[i]);
  }
}

// RoPE + repack: in fp32 [ROWS][nheads*128] -> out bf16 [B][nheads][S][128]
__global__ void rope_pack(const float* __restrict__ in, const float* __restrict__ cs,
                          const float* __restrict__ sn, ushort_t* __restrict__ out, int nheads){
  int idx = blockIdx.x*blockDim.x + threadIdx.x;   // ROWS*nheads*64 threads
  int dh  = idx & 63;
  int h   = (idx >> 6) % nheads;
  int row = idx / (64*nheads);
  int s   = row & (SEQ-1);
  int b   = row / SEQ;
  const float* r = in + (size_t)row*(nheads*D_HEAD) + h*D_HEAD;
  float t1 = r[dh], t2 = r[dh + 64];
  float c  = cs[s*64 + dh], si = sn[s*64 + dh];
  ushort_t* o = out + (((size_t)b*nheads + h)*SEQ + s)*D_HEAD;
  o[dh]      = f2bf(t1*c - t2*si);
  o[dh + 64] = f2bf(t2*c + t1*si);
}

// v fp32 [ROWS][512] -> vT bf16 [B][kv][128][S]  (d-major so P@V B-frags are contiguous)
__global__ void pack_vT(const float* __restrict__ vf, ushort_t* __restrict__ vT){
  int idx = blockIdx.x*blockDim.x + threadIdx.x;   // ROWS*KV_DIM threads
  int col = idx & (KV_DIM-1);
  int row = idx >> 9;
  int s   = row & (SEQ-1);
  int b   = row >> 11;
  int kv  = col >> 7, d = col & 127;
  vT[(((size_t)b*N_KV + kv)*D_HEAD + d)*SEQ + s] = f2bf(vf[idx]);
}

// ---------------- WMMA GEMM: C[M][N] f32 = A[M][K] bf16 * BT[N][K] bf16 ----------------
// block = 256 thr (8 waves). Wave tile 64x64 (16 WMMAs / 8 frag loads per k-step);
// block tile 128x256.
__global__ __launch_bounds__(256) void gemm_bf16_tn(const ushort_t* __restrict__ A,
                                                    const ushort_t* __restrict__ BT,
                                                    float* __restrict__ C,
                                                    int M, int N, int K){
  const int lane = threadIdx.x & 31;
  const int wave = threadIdx.x >> 5;
  const int lr   = lane & 15;
  const int m0 = blockIdx.y*128 + (wave & 1)*64;
  const int n0 = blockIdx.x*256 + (wave >> 1)*64;

  const ushort_t* ap[4];
  const ushort_t* bp[4];
  #pragma unroll
  for (int i = 0; i < 4; ++i) ap[i] = A  + (size_t)(m0 + 16*i + lr)*K;
  #pragma unroll
  for (int t = 0; t < 4; ++t) bp[t] = BT + (size_t)(n0 + 16*t + lr)*K;

  v8f acc[4][4];
  #pragma unroll
  for (int i = 0; i < 4; ++i)
    #pragma unroll
    for (int t = 0; t < 4; ++t) acc[i][t] = (v8f){0.f,0.f,0.f,0.f,0.f,0.f,0.f,0.f};

  for (int kk = 0; kk < K; kk += 32){
    v16bf fa[4];
    #pragma unroll
    for (int i = 0; i < 4; ++i) fa[i] = ldfragA(ap[i], kk, lane);
    #pragma unroll
    for (int t = 0; t < 4; ++t){
      v16bf fb = ldfragB(bp[t], kk, lane);
      #pragma unroll
      for (int i = 0; i < 4; ++i)
        acc[i][t] = wmma_bf16(fa[i], fb, acc[i][t]);
    }
  }

  const int mh = (lane >> 4) * 8;
  #pragma unroll
  for (int i = 0; i < 4; ++i)
    #pragma unroll
    for (int t = 0; t < 4; ++t)
      #pragma unroll
      for (int r = 0; r < 8; ++r){
        int m = m0 + 16*i + mh + r;
        int n = n0 + 16*t + lr;
        C[(size_t)m*N + n] = acc[i][t][r];
      }
}

// ---------------- Flash attention ----------------
// q: bf16 [B][16][S][128]; k: bf16 [B][4][S][128]; vT: bf16 [B][4][128][S]
// out: bf16 [ROWS][2048]. Block = 8 waves, wave owns 16 query rows,
// key tile = 64 (4 score sub-tiles) to amortize softmax/LDS-sync overhead.
__global__ __launch_bounds__(256) void flash_attn(const ushort_t* __restrict__ q,
                                                  const ushort_t* __restrict__ k,
                                                  const ushort_t* __restrict__ vT,
                                                  ushort_t* __restrict__ out){
  __shared__ ushort_t pbuf[8*16*64];   // per-wave 16x64 P tile (16 KB total)

  const int lane = threadIdx.x & 31;
  const int wave = threadIdx.x >> 5;
  const int lr   = lane & 15;
  const int mh   = (lane >> 4) * 8;
  const int b    = blockIdx.z;
  const int h    = blockIdx.y;
  const int kvh  = h >> 2;                      // h / N_REP
  const int qbase = blockIdx.x*128 + wave*16;

  // Q fragments (16 rows x 128 d), resident for the whole kernel
  const ushort_t* qrow = q + (((size_t)b*N_HEAD + h)*SEQ + qbase + lr)*D_HEAD;
  v16bf qa[4];
  #pragma unroll
  for (int d = 0; d < 4; ++d) qa[d] = ldfragA(qrow, d*32, lane);

  const ushort_t* kbp = k  + ((size_t)b*N_KV + kvh)*SEQ*D_HEAD;
  const ushort_t* vbp = vT + ((size_t)b*N_KV + kvh)*(size_t)D_HEAD*SEQ;

  v8f acc[8];
  float mrow[8], lrow[8];
  #pragma unroll
  for (int t = 0; t < 8; ++t) acc[t] = (v8f){0.f,0.f,0.f,0.f,0.f,0.f,0.f,0.f};
  #pragma unroll
  for (int r = 0; r < 8; ++r){ mrow[r] = -3.0e38f; lrow[r] = 0.f; }

  // logits pre-scaled by (1/sqrt(128)) * log2(e) so softmax runs on native exp2
  const float kscale = 0.12751744f;
  ushort_t* pwr = &pbuf[wave*1024];
  const int ntiles = (qbase + 79) >> 6;         // causal: keys 0..qbase+15, 64/tile

  for (int j = 0; j < ntiles; ++j){
    const int jb = j*64;

    // ---- S = q @ k^T  (four 16x16 tiles covering 64 keys) ----
    v8f st[4];
    #pragma unroll
    for (int half = 0; half < 4; ++half){
      v8f c = (v8f){0.f,0.f,0.f,0.f,0.f,0.f,0.f,0.f};
      const ushort_t* krow = kbp + (size_t)(jb + half*16 + lr)*D_HEAD;
      #pragma unroll
      for (int d = 0; d < 4; ++d){
        v16bf kf = ldfragB(krow, d*32, lane);
        c = wmma_bf16(qa[d], kf, c);
      }
      st[half] = c;
    }

    // ---- mask + online softmax (base-2 domain; row stats per (vgpr,lane-half)) ----
    #pragma unroll
    for (int r = 0; r < 8; ++r){
      const int qi = qbase + mh + r;
      float s[4];
      #pragma unroll
      for (int half = 0; half < 4; ++half){
        s[half] = st[half][r]*kscale;
        if (jb + half*16 + lr > qi) s[half] = -3.0e38f;
      }
      float vmax = fmaxf(fmaxf(s[0], s[1]), fmaxf(s[2], s[3]));
      #pragma unroll
      for (int msk = 1; msk < 16; msk <<= 1) vmax = fmaxf(vmax, __shfl_xor(vmax, msk, 16));
      const float mnew = fmaxf(mrow[r], vmax);
      float p[4], psum = 0.f;
      #pragma unroll
      for (int half = 0; half < 4; ++half){ p[half] = exp2f(s[half] - mnew); psum += p[half]; }
      #pragma unroll
      for (int msk = 1; msk < 16; msk <<= 1) psum += __shfl_xor(psum, msk, 16);
      const float corr = exp2f(mrow[r] - mnew);
      lrow[r] = lrow[r]*corr + psum;
      mrow[r] = mnew;
      #pragma unroll
      for (int t = 0; t < 8; ++t) acc[t][r] = acc[t][r]*corr;
      // stage P (C-layout -> LDS row-major) for the A-fragment reload
      const int m = mh + r;
      #pragma unroll
      for (int half = 0; half < 4; ++half)
        pwr[m*64 + half*16 + lr] = f2bf(p[half]);
    }
    asm volatile("s_wait_dscnt 0" ::: "memory");
    v16bf pA0 = ldfragA(pwr + lr*64,  0, lane);  // P keys 0..31 as A operand
    v16bf pA1 = ldfragA(pwr + lr*64, 32, lane);  // P keys 32..63 as A operand

    // ---- acc += P @ V  (8 n-tiles over d = 0..127, 2 k-chunks of 32 keys) ----
    #pragma unroll
    for (int t = 0; t < 8; ++t){
      const ushort_t* vrow = vbp + (size_t)(t*16 + lr)*SEQ + jb;
      v16bf vb0 = ldfragB(vrow,  0, lane);
      acc[t] = wmma_bf16(pA0, vb0, acc[t]);
      v16bf vb1 = ldfragB(vrow, 32, lane);
      acc[t] = wmma_bf16(pA1, vb1, acc[t]);
    }
  }

  // ---- epilogue: divide by l, write bf16 [ROWS][2048] for the O-proj GEMM ----
  const size_t obase = ((size_t)b*SEQ + qbase)*D_MODEL + (size_t)h*D_HEAD;
  #pragma unroll
  for (int r = 0; r < 8; ++r){
    const float inv = 1.0f / lrow[r];
    const int m = mh + r;
    #pragma unroll
    for (int t = 0; t < 8; ++t)
      out[obase + (size_t)m*D_MODEL + t*16 + lr] = f2bf(acc[t][r]*inv);
  }
}

// ---------------- launcher ----------------

extern "C" void kernel_launch(void* const* d_in, const int* in_sizes, int n_in,
                              void* d_out, int out_size, void* d_ws, size_t ws_size,
                              hipStream_t stream){
  const float* x    = (const float*)d_in[0];
  const float* cosp = (const float*)d_in[1];
  const float* sinp = (const float*)d_in[2];
  const float* wq   = (const float*)d_in[3];
  const float* wk   = (const float*)d_in[4];
  const float* wv   = (const float*)d_in[5];
  const float* wo   = (const float*)d_in[6];
  (void)in_sizes; (void)n_in; (void)out_size; (void)ws_size;

  const size_t MB = 1024*1024;
  unsigned char* ws = (unsigned char*)d_ws;
  // stage-1 regions (aliased later, stream order guarantees safety):
  ushort_t* xb   = (ushort_t*)(ws +  0*MB);   // 16 MB   (reused as qb)
  ushort_t* wqT  = (ushort_t*)(ws + 16*MB);   //  8 MB   (reused as kb)
  ushort_t* wkT  = (ushort_t*)(ws + 24*MB);   //  2 MB   (reused: vT spans 24..28)
  ushort_t* wvT  = (ushort_t*)(ws + 26*MB);   //  2 MB
  ushort_t* woT  = (ushort_t*)(ws + 28*MB);   //  8 MB   (persists to final GEMM)
  float*    qf   = (float*)   (ws + 36*MB);   // 32 MB
  float*    kf   = (float*)   (ws + 68*MB);   //  8 MB   (reused: attn spans 68..84)
  float*    vf   = (float*)   (ws + 76*MB);   //  8 MB
  ushort_t* qb   = xb;
  ushort_t* kb   = wqT;
  ushort_t* vT   = wkT;
  ushort_t* attn = (ushort_t*)(ws + 68*MB);   // 16 MB; total footprint: 84 MB

  const int THR = 256;

  // 1) precision convert + weight transposes
  f32_to_bf16_k   <<<(ROWS*D_MODEL)/THR, THR, 0, stream>>>(x,  xb,  ROWS*D_MODEL);
  transpose_to_bf16<<<(D_MODEL*D_MODEL)/THR, THR, 0, stream>>>(wq, wqT, D_MODEL, D_MODEL);
  transpose_to_bf16<<<(D_MODEL*KV_DIM )/THR, THR, 0, stream>>>(wk, wkT, D_MODEL, KV_DIM);
  transpose_to_bf16<<<(D_MODEL*KV_DIM )/THR, THR, 0, stream>>>(wv, wvT, D_MODEL, KV_DIM);
  transpose_to_bf16<<<(D_MODEL*D_MODEL)/THR, THR, 0, stream>>>(wo, woT, D_MODEL, D_MODEL);

  // 2) QKV projection GEMMs (WMMA)
  gemm_bf16_tn<<<dim3(D_MODEL/256, ROWS/128), THR, 0, stream>>>(xb, wqT, qf, ROWS, D_MODEL, D_MODEL);
  gemm_bf16_tn<<<dim3(KV_DIM /256, ROWS/128), THR, 0, stream>>>(xb, wkT, kf, ROWS, KV_DIM,  D_MODEL);
  gemm_bf16_tn<<<dim3(KV_DIM /256, ROWS/128), THR, 0, stream>>>(xb, wvT, vf, ROWS, KV_DIM,  D_MODEL);

  // 3) RoPE + head-major repack (q,k), d-major transpose repack (v)
  rope_pack<<<(ROWS*N_HEAD*64)/THR, THR, 0, stream>>>(qf, cosp, sinp, qb, N_HEAD);
  rope_pack<<<(ROWS*N_KV  *64)/THR, THR, 0, stream>>>(kf, cosp, sinp, kb, N_KV);
  pack_vT  <<<(ROWS*KV_DIM)   /THR, THR, 0, stream>>>(vf, vT);

  // 4) causal flash attention (WMMA for q@k^T and P@V)
  flash_attn<<<dim3(SEQ/128, N_HEAD, BATCH), THR, 0, stream>>>(qb, kb, vT, attn);

  // 5) output projection GEMM (WMMA) -> fp32 d_out
  gemm_bf16_tn<<<dim3(D_MODEL/256, ROWS/128), THR, 0, stream>>>(attn, woT, (float*)d_out,
                                                               ROWS, D_MODEL, D_MODEL);
}
